// Encoder_3624952398618
// MI455X (gfx1250) — matrix operand order repaired and checked by reading, not compiled, not run on hardware
//
#include <hip/hip_runtime.h>
#include <hip/hip_bf16.h>
#include <math.h>

typedef __attribute__((ext_vector_type(16))) _Float16 v16h;
typedef __attribute__((ext_vector_type(8)))  float    v8f;

#define SEQ      4096
#define HID      1024
#define G3H      3072
#define KT       32        // 32 k-tiles of 32 -> K=1024
#define NWG      96        // persistent workgroups (1 per WGP)
#define RPW      32        // weight rows per WG per matrix (3072/96)
#define PTHREADS 128       // 4 waves

// ---------------------------------------------------------------------------
// Kernel 1: gi0_all[S,3H] = embedding[input_ids] @ W_ih0^T + b_ih0   (WMMA f16)
// Each wave: one 16-row M tile x four 16-col N tiles, K=1024 in 32 WMMA steps.
// ---------------------------------------------------------------------------
__global__ __launch_bounds__(256)
void gi0_gemm_kernel(const int* __restrict__ ids,
                     const float* __restrict__ emb,
                     const float* __restrict__ Wih0,
                     const float* __restrict__ bih0,
                     float* __restrict__ gi0) {
  const int lane = threadIdx.x & 31;
  const int wv   = threadIdx.x >> 5;
  const int task = blockIdx.x * 8 + wv;    // 12288 wave tasks
  const int mt   = task / 48;              // 0..255  (M tile)
  const int ng   = task % 48;              // 0..47   (group of 4 N tiles)
  const int m0   = mt * 16;
  const int n0   = ng * 64;
  const int lm   = lane & 15;
  const int lh   = lane >> 4;

  // A-matrix gather: lane holds row M = lane%16 (both lane halves).
  const float* erow = emb + (long)ids[m0 + lm] * HID;

  v8f c0 = {}, c1 = {}, c2 = {}, c3 = {};
  for (int kk = 0; kk < KT; ++kk) {
    const int kb = kk * 32;
    // A in this code's contiguous k order: lanes 0-15 -> K 0..15; lanes 16-31 -> K 16..31 (matches B)
    v16h a;
    #pragma unroll
    for (int e = 0; e < 8; ++e) {
      a[e]     = (_Float16)erow[kb + lh * 16 + e];
      a[8 + e] = (_Float16)erow[kb + lh * 16 + 8 + e];
    }
    #pragma unroll
    for (int nt = 0; nt < 4; ++nt) {
      // B in the same contiguous k order as A: lane = N%16, lane>>4 selects K half; 16 contiguous K.
      const float* wrow = Wih0 + (long)(n0 + nt * 16 + lm) * HID + kb + lh * 16;
      v16h b;
      #pragma unroll
      for (int e = 0; e < 16; ++e) b[e] = (_Float16)wrow[e];
      v8f& c = (nt == 0) ? c0 : (nt == 1) ? c1 : (nt == 2) ? c2 : c3;
      c = __builtin_amdgcn_wmma_f32_16x16x32_f16(false, a, false, b, (short)0, c,
                                                 false, false);
    }
  }
  // C/D layout: VGPR r -> M=r (lanes 0-15) / M=8+r (lanes 16-31), N = lane%16.
  const int mbase = m0 + (lane >= 16 ? 8 : 0);
  #pragma unroll
  for (int nt = 0; nt < 4; ++nt) {
    const v8f  c    = (nt == 0) ? c0 : (nt == 1) ? c1 : (nt == 2) ? c2 : c3;
    const int  col  = n0 + nt * 16 + lm;
    const float bia = bih0[col];
    #pragma unroll
    for (int r = 0; r < 8; ++r)
      gi0[(long)(mbase + r) * G3H + col] = c[r] + bia;
  }
}

// ---------------------------------------------------------------------------
// Persistent GRU scan kernel
// ---------------------------------------------------------------------------
__device__ __forceinline__ float sigf(float x) { return 1.f / (1.f + __expf(-x)); }

// Grid-wide barrier: monotonic counter in L2, release-add / acquire-spin.
__device__ __forceinline__ void gbar(unsigned* ctr, unsigned barno) {
  __syncthreads();
  if (threadIdx.x == 0) {
    __hip_atomic_fetch_add(ctr, 1u, __ATOMIC_RELEASE, __HIP_MEMORY_SCOPE_AGENT);
    const unsigned target = barno * (unsigned)NWG;
    while (__hip_atomic_load(ctr, __ATOMIC_RELAXED, __HIP_MEMORY_SCOPE_AGENT) < target)
      __builtin_amdgcn_s_sleep(1);
  }
  __syncthreads();
  __builtin_amdgcn_fence(__ATOMIC_ACQUIRE, "agent");
}

// One wave: 16-output matvec tile, W fragments pre-swizzled in LDS, h in LDS (f32).
// Result: lanes 0-15, VGPR0 = out[n0 + lane].
__device__ __forceinline__ v8f wave_matvec(const _Float16* frag_base,
                                           const float* hv, int lane) {
  const int lm = lane & 15;
  const int lh = lane >> 4;
  v8f c = {};
  for (int kk = 0; kk < KT; ++kk) {
    v16h a = {};
    if (lm == 0) {  // only A-row 0 carries data (M=1 matvec)
      const float* hp = hv + kk * 32 + lh * 16;
      #pragma unroll
      for (int e = 0; e < 8; ++e) {
        a[e]     = (_Float16)hp[e];
        a[8 + e] = (_Float16)hp[8 + e];
      }
    }
    const v16h b = *(const v16h*)(frag_base + (size_t)(kk * 32 + lane) * 16);
    c = __builtin_amdgcn_wmma_f32_16x16x32_f16(false, a, false, b, (short)0, c,
                                               false, false);
  }
  return c;
}

__global__ __launch_bounds__(PTHREADS)
void gru_persistent_kernel(const float* __restrict__ hidden_init,
                           const float* __restrict__ Whh0, const float* __restrict__ bhh0,
                           const float* __restrict__ Wih1, const float* __restrict__ bih1,
                           const float* __restrict__ Whh1, const float* __restrict__ bhh1,
                           const float* __restrict__ Wout, const float* __restrict__ bout,
                           const float* __restrict__ gi0,
                           float* g_gh0, float* g_gh1, float* g_gi1,
                           unsigned* g_bar, float* out) {
  extern __shared__ char smem[];
  _Float16* wfrag = (_Float16*)smem;                 // 3 mats x 2 ntiles x 32 kk x 32 lanes x 16 = 192 KB
  float* h0  = (float*)(smem + 196608);              // 4 KB
  float* h1  = (float*)(smem + 196608 + 4096);       // 4 KB
  float* red = (float*)(smem + 196608 + 8192);       // 512 B

  const int tid  = threadIdx.x;
  const int lane = tid & 31;
  const int wv   = tid >> 5;
  const int r0   = blockIdx.x * RPW;                 // this WG's 32 output rows

  // ---- init hidden state (reference passes zeros, read anyway) ----
  for (int j = tid; j < HID; j += PTHREADS) {
    h0[j] = hidden_init[j];
    h1[j] = hidden_init[HID + j];
  }
  // ---- stage weight slices into LDS, pre-swizzled to B-fragment order ----
  const float* mats[3] = { Whh0, Whh1, Wih1 };
  for (int idx = tid; idx < 3 * 2 * KT * 32; idx += PTHREADS) {
    const int l  = idx & 31;            // fragment lane
    const int f  = idx >> 5;            // fragment index ((m*2+t)*32+kk)
    const int kk = f & 31;
    const int t  = (f >> 5) & 1;
    const int m  = f >> 6;
    const int row = r0 + t * 16 + (l & 15);
    const int kb  = kk * 32 + (l >> 4) * 16;
    const float* src = mats[m] + (long)row * HID + kb;
    _Float16*    dst = wfrag + (size_t)idx * 16;
    #pragma unroll
    for (int e = 0; e < 16; ++e) dst[e] = (_Float16)src[e];
  }
  __syncthreads();

  unsigned barno = 0;
  for (int s = 0; s < SEQ; ++s) {
    // ---- phase A: gh0 = Whh0@h0 (waves 0,1) and gh1 = Whh1@h1 (waves 2,3) ----
    {
      const int m = wv >> 1;
      const int t = wv & 1;
      const float*    hv   = (m == 0) ? h0 : h1;
      const _Float16* base = wfrag + (size_t)((m * 2 + t) * KT) * 32 * 16;
      const v8f c = wave_matvec(base, hv, lane);
      if (lane < 16) {
        const int row = r0 + t * 16 + lane;
        if (m == 0) g_gh0[row]                    = c[0] + bhh0[row];
        else        g_gh1[(s & 1) * G3H + row]    = c[0] + bhh1[row];
      }
    }
    gbar(g_bar, ++barno);

    // ---- phase B1: h0' computed redundantly by every WG (full H) ----
    const float* gi0s = gi0 + (size_t)s * G3H;
    for (int j = tid; j < HID; j += PTHREADS) {
      const float rr = sigf (gi0s[j]            + g_gh0[j]);
      const float zz = sigf (gi0s[HID + j]      + g_gh0[HID + j]);
      const float nn = tanhf(gi0s[2 * HID + j]  + rr * g_gh0[2 * HID + j]);
      h0[j] = (1.f - zz) * nn + zz * h0[j];
    }
    __syncthreads();

    // ---- phase B2: gi1 = Wih1@h0' (waves 0,1); waves 2,3 prefetch next gi0 ----
    if (wv < 2) {
      const _Float16* base = wfrag + (size_t)((2 * 2 + wv) * KT) * 32 * 16;
      const v8f c = wave_matvec(base, h0, lane);
      if (lane < 16) {
        const int row = r0 + wv * 16 + lane;
        g_gi1[row] = c[0] + bih1[row];
      }
    } else if (s + 1 < SEQ) {
      const float* nxt = gi0 + (size_t)(s + 1) * G3H;
      for (int j = (tid - 64) * 16; j < G3H; j += 64 * 16)
        __builtin_prefetch(nxt + j, 0, 0);   // -> global_prefetch_b8
    }
    gbar(g_bar, ++barno);

    // ---- phase B3: h1' (redundant per WG); gh1 ping-pong avoids 3rd barrier ----
    const float* gh1s = g_gh1 + (s & 1) * G3H;
    for (int j = tid; j < HID; j += PTHREADS) {
      const float rr = sigf (g_gi1[j]            + gh1s[j]);
      const float zz = sigf (g_gi1[HID + j]      + gh1s[HID + j]);
      const float nn = tanhf(g_gi1[2 * HID + j]  + rr * gh1s[2 * HID + j]);
      h1[j] = (1.f - zz) * nn + zz * h1[j];
    }
    __syncthreads();
  }

  // ---- epilogue: out = sigmoid(h1 . W_out + b_out), O==1, WG 0 only ----
  if (blockIdx.x == 0) {
    float acc = 0.f;
    for (int j = tid; j < HID; j += PTHREADS) acc += h1[j] * Wout[j];
    red[tid] = acc;
    __syncthreads();
    for (int off = PTHREADS / 2; off > 0; off >>= 1) {
      if (tid < off) red[tid] += red[tid + off];
      __syncthreads();
    }
    if (tid == 0) out[0] = sigf(red[0] + bout[0]);
  }
}

// ---------------------------------------------------------------------------
extern "C" void kernel_launch(void* const* d_in, const int* in_sizes, int n_in,
                              void* d_out, int out_size, void* d_ws, size_t ws_size,
                              hipStream_t stream) {
  (void)in_sizes; (void)n_in; (void)out_size; (void)ws_size;
  const int*   ids    = (const int*)  d_in[0];
  const float* hidden = (const float*)d_in[1];
  const float* emb    = (const float*)d_in[2];
  const float* Wih0   = (const float*)d_in[3];
  const float* Whh0   = (const float*)d_in[4];
  const float* bih0   = (const float*)d_in[5];
  const float* bhh0   = (const float*)d_in[6];
  const float* Wih1   = (const float*)d_in[7];
  const float* Whh1   = (const float*)d_in[8];
  const float* bih1   = (const float*)d_in[9];
  const float* bhh1   = (const float*)d_in[10];
  const float* Wout   = (const float*)d_in[11];
  const float* bout   = (const float*)d_in[12];

  char* ws = (char*)d_ws;
  float*    gi0   = (float*)ws;                                        // 50,331,648 B
  float*    g_gh0 = (float*)(ws + 50331648);                           // 12,288 B
  float*    g_gh1 = (float*)(ws + 50331648 + 12288);                   // 24,576 B (ping-pong)
  float*    g_gi1 = (float*)(ws + 50331648 + 12288 + 24576);           // 12,288 B
  unsigned* g_bar = (unsigned*)(ws + 50331648 + 12288 + 24576 + 12288);

  // Barrier counter must start at 0 every launch (graph-capture safe).
  hipMemsetAsync(g_bar, 0, 256, stream);

  // Batched input-gate GEMM (WMMA): 256 Mtiles * 48 Ngroups / 8 waves = 1536 blocks.
  gi0_gemm_kernel<<<1536, 256, 0, stream>>>(ids, emb, Wih0, bih0, gi0);

  // Persistent recurrent scan: 96 WGs x 128 threads, ~200.5 KB LDS each.
  const size_t lds_bytes = 196608 + 4096 + 4096 + 512;
  gru_persistent_kernel<<<NWG, PTHREADS, lds_bytes, stream>>>(
      hidden, Whh0, bhh0, Wih1, bih1, Whh1, bhh1, Wout, bout,
      gi0, g_gh0, g_gh1, g_gi1, g_bar, (float*)d_out);
}
